// NonLinearReadoutBlock_49091476194033
// MI455X (gfx1250) — matrix operand order repaired and checked by reading, not compiled, not run on hardware
//
#include <hip/hip_runtime.h>
#include <hip/hip_bf16.h>

typedef __attribute__((ext_vector_type(16))) _Float16     v16h;
typedef __attribute__((ext_vector_type(8)))  float        v8f;
typedef __attribute__((ext_vector_type(4)))  unsigned int v4u;
typedef __attribute__((ext_vector_type(4)))  float        v4f;

#define N_NODES    65536
#define DIM_IN     4096
#define K_DIM      1024
#define N_DIM      1024
#define TILE_M     128
#define LDS_STRIDE 1032   /* 1024 + 8 halves padding -> bank-spread rows */
#define SMEM_BYTES (TILE_M * LDS_STRIDE * sizeof(_Float16))   /* 264192 B */

// ---------------------------------------------------------------------------
// Pack W1 (f32, [K=1024][N=1024] row-major) into f16 B-fragments for
// v_wmma_f32_16x16x32_f16: fragment (n16, k32), lane l holds 16 halves:
//   col = n16*16 + (l&15), K = k32*32 + (l<16 ? 0 : 16) + j  (j = 0..15)
// Stored contiguously: W1h[((n16*32 + k32)*32 + lane)*16 + j]  (2 MB total).
// ---------------------------------------------------------------------------
__global__ void pack_w1_f16(const float* __restrict__ W1,
                            _Float16* __restrict__ W1h) {
  const int t    = blockIdx.x * blockDim.x + threadIdx.x;  // 65536 threads
  const int lane = t & 31;
  const int frag = t >> 5;          // n16*32 + k32
  const int k32  = frag & 31;
  const int n16  = frag >> 5;
  const int col  = n16 * 16 + (lane & 15);
  const int kb   = k32 * 32 + ((lane & 16) ? 16 : 0);
  _Float16* dst  = W1h + ((size_t)frag * 32 + lane) * 16;
#pragma unroll
  for (int j = 0; j < 16; ++j)
    dst[j] = (_Float16)W1[(size_t)(kb + j) * N_DIM + col];
}

// ---------------------------------------------------------------------------
// Fused: out = (silu(x[:, :1024] @ W1 / 32) * 1.679) @ W2 / 32
// One block = 128 rows. x tile staged as f16 in LDS; each wave owns 128 N
// columns and all 8 M-tiles; h never leaves registers.
// ---------------------------------------------------------------------------
__global__ __launch_bounds__(256) void fused_mlp_readout(
    const float* __restrict__ x, const _Float16* __restrict__ W1h,
    const float* __restrict__ W2, float* __restrict__ out) {
  extern __shared__ __align__(16) char smem_raw[];
  _Float16* smem = (_Float16*)smem_raw;

  const int    tid     = threadIdx.x;
  const int    lane    = tid & 31;
  const int    wave    = tid >> 5;
  const size_t rowBase = (size_t)blockIdx.x * TILE_M;

  // ---- Stage x[:, :1024] tile -> f16 in LDS (coalesced float4 loads) ----
  for (int i = tid; i < TILE_M * 256; i += 256) {   // 256 float4 per row
    const int r  = i >> 8;
    const int c4 = i & 255;
    const v4f v = *(const v4f*)(x + (rowBase + r) * DIM_IN + (size_t)c4 * 4);
    union { _Float16 h[4]; unsigned long long u; } p;
    p.h[0] = (_Float16)v.x; p.h[1] = (_Float16)v.y;
    p.h[2] = (_Float16)v.z; p.h[3] = (_Float16)v.w;
    *(unsigned long long*)(smem + (size_t)r * LDS_STRIDE + c4 * 4) = p.u;
  }
  __syncthreads();

  const float RS1      = 0.03125f;            // 1/sqrt(1024)
  const float RS2_NORM = 0.03125f * 1.679f;   // (1/sqrt(1024)) * SILU_NORM

  float oacc[8][8];
#pragma unroll
  for (int m = 0; m < 8; ++m)
#pragma unroll
    for (int v = 0; v < 8; ++v) oacc[m][v] = 0.f;

  const int r0  = lane & 15;              // A/B: row-in-tile / col-in-tile
  const int off = (lane & 16) ? 16 : 0;   // A layout: upper lanes -> K+16

  // A-fragment base pointers: invariant across the whole nt/k loop nest.
  // Hoisting them keeps inner-loop LDS addressing to base + small offset.
  const _Float16* aptr[8];
#pragma unroll
  for (int m = 0; m < 8; ++m)
    aptr[m] = smem + (size_t)(m * 16 + r0) * LDS_STRIDE + off;

  for (int nt = 0; nt < 8; ++nt) {
    const int n16 = wave * 8 + nt;        // N-tile index, 0..63

    v8f C[8];
#pragma unroll
    for (int m = 0; m < 8; ++m)
#pragma unroll
      for (int e = 0; e < 8; ++e) C[m][e] = 0.f;

    const _Float16* bptr = W1h + ((size_t)n16 * 1024 + lane) * 16;
#pragma unroll 2
    for (int k32 = 0; k32 < 32; ++k32) {
      const v16h b = *(const v16h*)bptr;  // 32B coalesced B fragment (L2)
      __builtin_prefetch(bptr + 1024, 0, 1);  // 2 k-steps ahead -> global_prefetch
      bptr += 512;                        // next k32 fragment

#pragma unroll
      for (int m = 0; m < 8; ++m) {
        // A fragment (contiguous-K 16x32 layout, k = 16h + s as in W1h): two 16B LDS chunks,
        // K = kb+off+0..7 and K = kb+off+8..15, row = m*16 + (lane&15)
        const _Float16* ap = aptr[m] + k32 * 32;
        union { v4u q[2]; v16h h; } af;
        af.q[0] = *(const v4u*)(ap);
        af.q[1] = *(const v4u*)(ap + 8);
        C[m] = __builtin_amdgcn_wmma_f32_16x16x32_f16(
                   false, af.h, false, b, (short)0, C[m], false, false);
      }
    }

    // ---- epilogue: scale, silu, dot with W2 column weight, accumulate ----
    const float w2v = W2[n16 * 16 + r0] * RS2_NORM;  // C layout: N = lane&15
#pragma unroll
    for (int m = 0; m < 8; ++m)
#pragma unroll
      for (int v = 0; v < 8; ++v) {
        const float h = C[m][v] * RS1;
        const float s = h / (1.f + __expf(-h));      // silu
        oacc[m][v] += s * w2v;
      }
  }

  // ---- reduce the 16 lanes sharing each M row, write 1 float per row ----
#pragma unroll
  for (int m = 0; m < 8; ++m)
#pragma unroll
    for (int v = 0; v < 8; ++v) {
      float s = oacc[m][v];
      s += __shfl_xor(s, 1, 32);
      s += __shfl_xor(s, 2, 32);
      s += __shfl_xor(s, 4, 32);
      s += __shfl_xor(s, 8, 32);
      if (r0 == 0)   // lanes 0 (rows m*16+v) and 16 (rows m*16+8+v)
        out[rowBase + m * 16 + ((lane >> 4) << 3) + v] = s;
    }
}

extern "C" void kernel_launch(void* const* d_in, const int* in_sizes, int n_in,
                              void* d_out, int out_size, void* d_ws, size_t ws_size,
                              hipStream_t stream) {
  const float* x   = (const float*)d_in[0];   // [65536, 4096] f32
  const float* W1  = (const float*)d_in[1];   // [1024, 1024] f32
  const float* W2  = (const float*)d_in[2];   // [1024, 1] f32
  float*       out = (float*)d_out;           // [65536] f32
  _Float16*    W1h = (_Float16*)d_ws;         // 2 MB packed B fragments

  (void)hipFuncSetAttribute((const void*)fused_mlp_readout,
                            hipFuncAttributeMaxDynamicSharedMemorySize,
                            (int)SMEM_BYTES);

  pack_w1_f16<<<(64 * 32 * 32) / 256, 256, 0, stream>>>(W1, W1h);
  fused_mlp_readout<<<N_NODES / TILE_M, 256, SMEM_BYTES, stream>>>(x, W1h, W2, out);
}
